// Decoder_71786083386060
// MI455X (gfx1250) — matrix operand order repaired and checked by reading, not compiled, not run on hardware
//
#include <hip/hip_runtime.h>

// ---------------------------------------------------------------------------
// GPT-2-style decoder forward for MI455X (gfx1250, wave32, WMMA bf16).
// All GEMMs run through v_wmma_f32_16x16x32_bf16 (fp32 accumulate).
// Wave tile is 32xNT*16 (NT=4 for large-N GEMMs -> 8 WMMA / 12 b128 loads).
// ---------------------------------------------------------------------------

typedef __attribute__((ext_vector_type(16))) __bf16 v16bf;
typedef __attribute__((ext_vector_type(8)))  __bf16 v8bf;
typedef __attribute__((ext_vector_type(8)))  float  v8f;

namespace {
constexpr int V_  = 32000;
constexpr int D_  = 1024;
constexpr int T_  = 1024;
constexpr int B_  = 2;
constexpr int H_  = 16;
constexpr int L_  = 2;
constexpr int FF_ = 4096;
constexpr int HD_ = D_ / H_;      // 64
constexpr int BT_ = B_ * T_;      // 2048
}

enum : int {
  EPI_BF16           = 0,  // Cb = (bf16)acc
  EPI_SCALE_F32      = 1,  // Cf = acc * scale
  EPI_BIAS_RES_F32   = 2,  // Cf = acc + bias[col] + res[row,col]
  EPI_BIAS_RELU_BF16 = 3,  // Cb = (bf16)relu(acc + bias[col])
  EPI_BIAS_F32       = 4   // Cf = acc + bias[col]
};

// ---------------------------------------------------------------------------
// Batched GEMM: C[M,N] = A[M,K] (bf16 row-major, ld=lda) * Bt[N,K]^T (bf16
// row-major, ld=ldb). Block = 256 threads = 8 waves, wave grid 4(M) x 2(N).
// Each wave owns a 32 x (NT*16) tile (2 x NT WMMA 16x16 tiles).
// Block tile = 128 x (32*NT). grid.z batches over (b,h); per-batch base
// offsets are zb*OffB + zh*OffH. M % 128 == 0, N % (32*NT) == 0, K % 32 == 0
// are guaranteed by the caller.
// ---------------------------------------------------------------------------
template <int EPI, int NT>
__global__ __launch_bounds__(256) void gemm_bt_kernel(
    const __bf16* __restrict__ A, const __bf16* __restrict__ Bt,
    float* __restrict__ Cf, __bf16* __restrict__ Cb,
    const float* __restrict__ bias, const float* __restrict__ res,
    int K, int lda, int ldb, int ldc,
    long aOffB, long aOffH, long bOffB, long bOffH,
    long cOffB, long cOffH, int Hdiv, float scale)
{
  const int lane  = threadIdx.x & 31;
  const int wave  = threadIdx.x >> 5;
  const int tileM = blockIdx.x * 128 + (wave & 3) * 32;
  const int tileN = blockIdx.y * (32 * NT) + (wave >> 2) * (16 * NT);
  const int z  = blockIdx.z;
  const int zb = z / Hdiv;
  const int zh = z - zb * Hdiv;
  A  += (long)zb * aOffB + (long)zh * aOffH;
  Bt += (long)zb * bOffB + (long)zh * bOffH;
  const long cbase = (long)zb * cOffB + (long)zh * cOffH;

  // Fragment addressing (contiguous K order, same for A and B):
  //  A 16x32: lane holds row (lane&15); lanes 0-15 K=0..15,
  //           lanes 16-31 K=16..31  -> two 16B chunks.
  //  B 32x16: lane holds col (lane&15); lanes 0-15 K=0..15, 16-31 K=16..31.
  const int mrow = lane & 15;
  const int hi   = lane >> 4;     // 0 or 1
  const int akc  = hi * 16;       // A K-chunk base
  const int bkc  = hi * 16;       // B K base

  const __bf16* a0p = A + (long)(tileM + mrow) * lda + akc;
  const __bf16* a1p = a0p + 16 * lda;
  const __bf16* bp[NT];
#pragma unroll
  for (int j = 0; j < NT; ++j)
    bp[j] = Bt + (long)(tileN + j * 16 + mrow) * ldb + bkc;

  v8f acc[2][NT];
#pragma unroll
  for (int j = 0; j < NT; ++j) { acc[0][j] = {}; acc[1][j] = {}; }

  for (int k = 0; k < K; k += 32) {
    // WGP-scope prefetch of fragments 4 K-steps ahead (global_prefetch_b8,
    // pulls into all cache levels). Branchless in-bounds clamp: when the
    // lookahead would run past K, re-prefetch the current (valid) chunk.
    const int kpf = (k + 128 < K) ? (k + 128) : k;
    __builtin_prefetch(a0p + kpf, 0, 3);
    __builtin_prefetch(a1p + kpf, 0, 3);
    __builtin_prefetch(bp[0] + kpf, 0, 3);

    v8bf a0lo = *(const v8bf*)(a0p + k);
    v8bf a0hi = *(const v8bf*)(a0p + k + 8);
    v8bf a1lo = *(const v8bf*)(a1p + k);
    v8bf a1hi = *(const v8bf*)(a1p + k + 8);
    v16bf a0 = __builtin_shufflevector(a0lo, a0hi,
        0,1,2,3,4,5,6,7,8,9,10,11,12,13,14,15);
    v16bf a1 = __builtin_shufflevector(a1lo, a1hi,
        0,1,2,3,4,5,6,7,8,9,10,11,12,13,14,15);
    v16bf bf[NT];
#pragma unroll
    for (int j = 0; j < NT; ++j) bf[j] = *(const v16bf*)(bp[j] + k);
#pragma unroll
    for (int j = 0; j < NT; ++j)
      acc[0][j] = __builtin_amdgcn_wmma_f32_16x16x32_bf16(
          false, a0, false, bf[j], (short)0, acc[0][j], false, false);
#pragma unroll
    for (int j = 0; j < NT; ++j)
      acc[1][j] = __builtin_amdgcn_wmma_f32_16x16x32_bf16(
          false, a1, false, bf[j], (short)0, acc[1][j], false, false);
  }

  // C/D layout: lanes 0-15: N=lane, M=r; lanes 16-31: N=lane-16, M=8+r.
  const int nlo = lane & 15;
  const int mhi = hi * 8;
#pragma unroll
  for (int mi = 0; mi < 2; ++mi) {
#pragma unroll
    for (int nj = 0; nj < NT; ++nj) {
#pragma unroll
      for (int r = 0; r < 8; ++r) {
        const int row = tileM + mi * 16 + mhi + r;
        const int col = tileN + nj * 16 + nlo;
        float v = acc[mi][nj][r];
        if constexpr (EPI == EPI_SCALE_F32) v *= scale;
        if constexpr (EPI == EPI_BIAS_RES_F32 || EPI == EPI_BIAS_RELU_BF16 ||
                      EPI == EPI_BIAS_F32) v += bias[col];
        if constexpr (EPI == EPI_BIAS_RES_F32) v += res[(long)row * ldc + col];
        if constexpr (EPI == EPI_BIAS_RELU_BF16) v = fmaxf(v, 0.0f);
        const long off = cbase + (long)row * ldc + col;
        if constexpr (EPI == EPI_BF16 || EPI == EPI_BIAS_RELU_BF16)
          Cb[off] = (__bf16)v;
        else
          Cf[off] = v;
      }
    }
  }
  (void)scale; (void)bias; (void)res; (void)Cf; (void)Cb;
}

// ---------------------------------------------------------------------------
// x[b,t,:] = word_emb[idx[b,t]] + pos_emb[t]
// ---------------------------------------------------------------------------
__global__ __launch_bounds__(256) void embed_kernel(
    const float* __restrict__ we, const float* __restrict__ pe,
    const int* __restrict__ idx, float* __restrict__ x)
{
  const long i = (long)blockIdx.x * 256 + threadIdx.x;
  if (i >= (long)BT_ * D_) return;
  const int d  = (int)(i & (D_ - 1));
  const long bt = i >> 10;                  // D_ == 1024
  const int t  = (int)(bt & (T_ - 1));
  const int tok = idx[bt];
  x[i] = we[(long)tok * D_ + d] + pe[(long)t * D_ + d];
}

// ---------------------------------------------------------------------------
// Row LayerNorm over D; writes bf16 and/or fp32 outputs.
// ---------------------------------------------------------------------------
__global__ __launch_bounds__(256) void layernorm_kernel(
    const float* __restrict__ x, const float* __restrict__ g,
    const float* __restrict__ b, __bf16* __restrict__ ob,
    float* __restrict__ of)
{
  const int row = blockIdx.x;
  const float* xr = x + (long)row * D_;
  __shared__ float s1[256], s2[256];
  float sum = 0.f, sq = 0.f;
  for (int i = threadIdx.x; i < D_; i += 256) {
    float v = xr[i]; sum += v; sq += v * v;
  }
  s1[threadIdx.x] = sum; s2[threadIdx.x] = sq; __syncthreads();
  for (int st = 128; st > 0; st >>= 1) {
    if (threadIdx.x < st) { s1[threadIdx.x] += s1[threadIdx.x + st];
                            s2[threadIdx.x] += s2[threadIdx.x + st]; }
    __syncthreads();
  }
  const float mean = s1[0] * (1.0f / D_);
  const float var  = s2[0] * (1.0f / D_) - mean * mean;
  const float rs   = rsqrtf(var + 1e-5f);
  for (int i = threadIdx.x; i < D_; i += 256) {
    float v = (xr[i] - mean) * rs * g[i] + b[i];
    if (ob) ob[(long)row * D_ + i] = (__bf16)v;
    if (of) of[(long)row * D_ + i] = v;
  }
}

// ---------------------------------------------------------------------------
// Causal softmax of one (b,h,q) row of length T, in place on fp32 scores
// (the maps output region), plus a bf16 copy for the PV WMMA GEMM.
// ---------------------------------------------------------------------------
__global__ __launch_bounds__(256) void softmax_causal_kernel(
    float* __restrict__ w, __bf16* __restrict__ wb)
{
  const int q  = blockIdx.x;
  const int bh = blockIdx.y;
  float* wr = w + ((long)bh * T_ + q) * T_;
  __bf16* wbr = wb + ((long)bh * T_ + q) * T_;
  __shared__ float red[256];
  float mx = -__builtin_inff();
  for (int i = threadIdx.x; i <= q; i += 256) mx = fmaxf(mx, wr[i]);
  red[threadIdx.x] = mx; __syncthreads();
  for (int st = 128; st > 0; st >>= 1) {
    if (threadIdx.x < st)
      red[threadIdx.x] = fmaxf(red[threadIdx.x], red[threadIdx.x + st]);
    __syncthreads();
  }
  mx = red[0]; __syncthreads();
  float sum = 0.f;
  for (int i = threadIdx.x; i <= q; i += 256) sum += __expf(wr[i] - mx);
  red[threadIdx.x] = sum; __syncthreads();
  for (int st = 128; st > 0; st >>= 1) {
    if (threadIdx.x < st) red[threadIdx.x] += red[threadIdx.x + st];
    __syncthreads();
  }
  const float inv = 1.0f / red[0];
  for (int i = threadIdx.x; i < T_; i += 256) {
    float p = (i <= q) ? __expf(wr[i] - mx) * inv : 0.0f;
    wr[i]  = p;
    wbr[i] = (__bf16)p;
  }
}

// ---------------------------------------------------------------------------
// vT[b,h,d,t] = v[b,t, h*HD + d]  (bf16 -> bf16), so PV fits A*B^T form.
// ---------------------------------------------------------------------------
__global__ __launch_bounds__(256) void transpose_v_kernel(
    const __bf16* __restrict__ v, __bf16* __restrict__ vT)
{
  const long i = (long)blockIdx.x * 256 + threadIdx.x;
  if (i >= (long)B_ * H_ * HD_ * T_) return;
  const int t  = (int)(i & (T_ - 1));
  const long r = i >> 10;
  const int d  = (int)(r & (HD_ - 1));
  const long r2 = r >> 6;
  const int h  = (int)(r2 & (H_ - 1));
  const int b  = (int)(r2 >> 4);
  vT[i] = v[((long)(b * T_ + t)) * D_ + h * HD_ + d];
}

// ---------------------------------------------------------------------------
// Wt[n,k] = (bf16)W[k,n]  — fp32 [K,N] -> transposed bf16 [N,K].
// ---------------------------------------------------------------------------
__global__ __launch_bounds__(256) void convt_kernel(
    const float* __restrict__ W, __bf16* __restrict__ Wt, int K, int N)
{
  const long i = (long)blockIdx.x * 256 + threadIdx.x;
  if (i >= (long)K * N) return;
  const long n = i / K;
  const int  k = (int)(i - n * K);
  Wt[i] = (__bf16)W[(long)k * N + n];
}

// ---------------------------------------------------------------------------
// Log-softmax of each row (length n) -> out (used for `probs`).
// ---------------------------------------------------------------------------
__global__ __launch_bounds__(256) void logsoftmax_rows_kernel(
    const float* __restrict__ in, float* __restrict__ out, int n)
{
  const int row = blockIdx.x;
  const float* r = in + (long)row * n;
  __shared__ float red[256];
  float mx = -__builtin_inff();
  for (int i = threadIdx.x; i < n; i += 256) mx = fmaxf(mx, r[i]);
  red[threadIdx.x] = mx; __syncthreads();
  for (int st = 128; st > 0; st >>= 1) {
    if (threadIdx.x < st)
      red[threadIdx.x] = fmaxf(red[threadIdx.x], red[threadIdx.x + st]);
    __syncthreads();
  }
  mx = red[0]; __syncthreads();
  float sum = 0.f;
  for (int i = threadIdx.x; i < n; i += 256) sum += __expf(r[i] - mx);
  red[threadIdx.x] = sum; __syncthreads();
  for (int st = 128; st > 0; st >>= 1) {
    if (threadIdx.x < st) red[threadIdx.x] += red[threadIdx.x + st];
    __syncthreads();
  }
  const float lse = mx + __logf(red[0]);
  for (int i = threadIdx.x; i < n; i += 256)
    out[(long)row * n + i] = r[i] - lse;
}

// ---------------------------------------------------------------------------
// Cross-entropy: loss += -(logits[row,tgt] - logsumexp(row)) / rows.
// ---------------------------------------------------------------------------
__global__ __launch_bounds__(256) void loss_kernel(
    const float* __restrict__ logits, const int* __restrict__ targets,
    float* __restrict__ loss, int n, float invRows)
{
  const int row = blockIdx.x;
  const float* r = logits + (long)row * n;
  __shared__ float red[256];
  float mx = -__builtin_inff();
  for (int i = threadIdx.x; i < n; i += 256) mx = fmaxf(mx, r[i]);
  red[threadIdx.x] = mx; __syncthreads();
  for (int st = 128; st > 0; st >>= 1) {
    if (threadIdx.x < st)
      red[threadIdx.x] = fmaxf(red[threadIdx.x], red[threadIdx.x + st]);
    __syncthreads();
  }
  mx = red[0]; __syncthreads();
  float sum = 0.f;
  for (int i = threadIdx.x; i < n; i += 256) sum += __expf(r[i] - mx);
  red[threadIdx.x] = sum; __syncthreads();
  for (int st = 128; st > 0; st >>= 1) {
    if (threadIdx.x < st) red[threadIdx.x] += red[threadIdx.x + st];
    __syncthreads();
  }
  if (threadIdx.x == 0) {
    const float lse = mx + __logf(red[0]);
    const float lp  = r[targets[row]] - lse;
    atomicAdd(loss, -lp * invRows);
  }
}

// ---------------------------------------------------------------------------
// Host-side launch helper for the GEMM template.
// ---------------------------------------------------------------------------
template <int EPI, int NT>
static void launch_gemm(hipStream_t s, const __bf16* A, const __bf16* Bt,
                        float* Cf, __bf16* Cb, const float* bias,
                        const float* res, int M, int N, int K,
                        int lda, int ldb, int ldc,
                        int batch, int Hdiv,
                        long aOffB, long aOffH, long bOffB, long bOffH,
                        long cOffB, long cOffH, float scale)
{
  dim3 grid(M / 128, N / (32 * NT), batch);
  gemm_bt_kernel<EPI, NT><<<grid, 256, 0, s>>>(
      A, Bt, Cf, Cb, bias, res, K, lda, ldb, ldc,
      aOffB, aOffH, bOffB, bOffH, cOffB, cOffH, Hdiv, scale);
}

extern "C" void kernel_launch(void* const* d_in, const int* in_sizes, int n_in,
                              void* d_out, int out_size, void* d_ws,
                              size_t ws_size, hipStream_t stream)
{
  (void)in_sizes; (void)n_in; (void)out_size; (void)ws_size;
  const int*   idx      = (const int*)d_in[0];
  const int*   targets  = (const int*)d_in[1];
  const float* word_emb = (const float*)d_in[2];
  const float* pos_emb  = (const float*)d_in[3];
  const float* ln1_g    = (const float*)d_in[4];
  const float* ln1_b    = (const float*)d_in[5];
  const float* wq       = (const float*)d_in[6];
  const float* wk       = (const float*)d_in[7];
  const float* wv       = (const float*)d_in[8];
  const float* proj_w   = (const float*)d_in[9];
  const float* proj_b   = (const float*)d_in[10];
  const float* ln2_g    = (const float*)d_in[11];
  const float* ln2_b    = (const float*)d_in[12];
  const float* ff_w1    = (const float*)d_in[13];
  const float* ff_b1    = (const float*)d_in[14];
  const float* ff_w2    = (const float*)d_in[15];
  const float* ff_b2    = (const float*)d_in[16];
  const float* lnf_g    = (const float*)d_in[17];
  const float* lnf_b    = (const float*)d_in[18];
  const float* lm_w     = (const float*)d_in[19];
  const float* lm_b     = (const float*)d_in[20];

  // Output layout: logits [BT,V] | loss [1] | maps [L,B,H,T,T] | probs [B,T,D]
  float* out    = (float*)d_out;
  float* logits = out;
  float* loss   = out + (size_t)BT_ * V_;
  float* maps   = loss + 1;
  float* probs  = maps + (size_t)L_ * B_ * H_ * T_ * T_;

  // Workspace bump allocator (256B aligned).
  char* p = (char*)d_ws;
  auto alloc = [&](size_t bytes) -> char* {
    char* r = p;
    p += (bytes + 255) & ~(size_t)255;
    return r;
  };
  float*  x      = (float*) alloc((size_t)BT_ * D_ * 4);
  __bf16* h_bf   = (__bf16*)alloc((size_t)BT_ * D_ * 2);
  __bf16* q_bf   = (__bf16*)alloc((size_t)BT_ * D_ * 2);
  __bf16* k_bf   = (__bf16*)alloc((size_t)BT_ * D_ * 2);
  __bf16* v_bf   = (__bf16*)alloc((size_t)BT_ * D_ * 2);
  __bf16* vT_bf  = (__bf16*)alloc((size_t)B_ * H_ * HD_ * T_ * 2);
  __bf16* wei_bf = (__bf16*)alloc((size_t)B_ * H_ * T_ * T_ * 2);
  __bf16* att_bf = (__bf16*)alloc((size_t)BT_ * D_ * 2);
  __bf16* ff1_bf = (__bf16*)alloc((size_t)BT_ * FF_ * 2);
  float*  xf     = (float*) alloc((size_t)BT_ * D_ * 4);
  __bf16* xf_bf  = (__bf16*)alloc((size_t)BT_ * D_ * 2);
  __bf16* wT     = (__bf16*)alloc((size_t)V_ * D_ * 2);  // reused for all weights

  const float attScale = 0.03125f;  // D^-0.5, D = 1024
  const long  nElemX   = (long)BT_ * D_;
  const int   blkX     = (int)((nElemX + 255) / 256);

  // x = word_emb[idx] + pos_emb
  embed_kernel<<<blkX, 256, 0, stream>>>(word_emb, pos_emb, idx, x);

  for (int l = 0; l < L_; ++l) {
    const float* wq_l = wq + (size_t)l * D_ * D_;
    const float* wk_l = wk + (size_t)l * D_ * D_;
    const float* wv_l = wv + (size_t)l * D_ * D_;
    const float* pw_l = proj_w + (size_t)l * D_ * D_;
    const float* pb_l = proj_b + (size_t)l * D_;
    const float* w1_l = ff_w1 + (size_t)l * D_ * FF_;
    const float* b1_l = ff_b1 + (size_t)l * FF_;
    const float* w2_l = ff_w2 + (size_t)l * FF_ * D_;
    const float* b2_l = ff_b2 + (size_t)l * D_;
    float* maps_l = maps + (size_t)l * B_ * H_ * T_ * T_;

    // h = LN1(x)  (bf16)
    layernorm_kernel<<<BT_, 256, 0, stream>>>(
        x, ln1_g + (size_t)l * D_, ln1_b + (size_t)l * D_, h_bf, nullptr);

    // q/k/v = h @ W  (bf16 WMMA, bf16 out)
    const int blkW_DD = (int)(((long)D_ * D_ + 255) / 256);
    convt_kernel<<<blkW_DD, 256, 0, stream>>>(wq_l, wT, D_, D_);
    launch_gemm<EPI_BF16, 4>(stream, h_bf, wT, nullptr, q_bf, nullptr, nullptr,
                             BT_, D_, D_, D_, D_, D_, 1, 1, 0,0,0,0,0,0, 1.f);
    convt_kernel<<<blkW_DD, 256, 0, stream>>>(wk_l, wT, D_, D_);
    launch_gemm<EPI_BF16, 4>(stream, h_bf, wT, nullptr, k_bf, nullptr, nullptr,
                             BT_, D_, D_, D_, D_, D_, 1, 1, 0,0,0,0,0,0, 1.f);
    convt_kernel<<<blkW_DD, 256, 0, stream>>>(wv_l, wT, D_, D_);
    launch_gemm<EPI_BF16, 4>(stream, h_bf, wT, nullptr, v_bf, nullptr, nullptr,
                             BT_, D_, D_, D_, D_, D_, 1, 1, 0,0,0,0,0,0, 1.f);

    // scores = q @ k^T * D^-0.5, batched over (b,h), written into maps
    launch_gemm<EPI_SCALE_F32, 4>(stream, q_bf, k_bf, maps_l, nullptr, nullptr,
                                  nullptr, T_, T_, HD_, D_, D_, T_,
                                  B_ * H_, H_,
                                  (long)T_ * D_, (long)HD_,
                                  (long)T_ * D_, (long)HD_,
                                  (long)H_ * T_ * T_, (long)T_ * T_, attScale);

    // causal softmax in place (fp32 maps) + bf16 copy for PV GEMM
    softmax_causal_kernel<<<dim3(T_, B_ * H_), 256, 0, stream>>>(maps_l, wei_bf);

    // vT for A*B^T form of PV
    transpose_v_kernel<<<(int)(((long)B_ * H_ * HD_ * T_ + 255) / 256), 256, 0,
                         stream>>>(v_bf, vT_bf);

    // att[b,t, h*HD+d] = wei @ v   (batched over (b,h), bf16 out; N=64 -> NT=2)
    launch_gemm<EPI_BF16, 2>(stream, wei_bf, vT_bf, nullptr, att_bf, nullptr,
                             nullptr, T_, HD_, T_, T_, T_, D_,
                             B_ * H_, H_,
                             (long)H_ * T_ * T_, (long)T_ * T_,
                             (long)H_ * HD_ * T_, (long)HD_ * T_,
                             (long)T_ * D_, (long)HD_, 1.f);

    // x = x + att @ proj_w + proj_b
    convt_kernel<<<blkW_DD, 256, 0, stream>>>(pw_l, wT, D_, D_);
    launch_gemm<EPI_BIAS_RES_F32, 4>(stream, att_bf, wT, x, nullptr, pb_l, x,
                                     BT_, D_, D_, D_, D_, D_, 1, 1,
                                     0,0,0,0,0,0, 1.f);

    // h2 = LN2(x)
    layernorm_kernel<<<BT_, 256, 0, stream>>>(
        x, ln2_g + (size_t)l * D_, ln2_b + (size_t)l * D_, h_bf, nullptr);

    // r = relu(h2 @ ff_w1 + ff_b1)  (bf16)
    const int blkW_DF = (int)(((long)D_ * FF_ + 255) / 256);
    convt_kernel<<<blkW_DF, 256, 0, stream>>>(w1_l, wT, D_, FF_);
    launch_gemm<EPI_BIAS_RELU_BF16, 4>(stream, h_bf, wT, nullptr, ff1_bf, b1_l,
                                       nullptr, BT_, FF_, D_, D_, D_, FF_, 1, 1,
                                       0,0,0,0,0,0, 1.f);

    // x = x + r @ ff_w2 + ff_b2
    convt_kernel<<<blkW_DF, 256, 0, stream>>>(w2_l, wT, FF_, D_);
    launch_gemm<EPI_BIAS_RES_F32, 4>(stream, ff1_bf, wT, x, nullptr, b2_l, x,
                                     BT_, D_, FF_, FF_, FF_, D_, 1, 1,
                                     0,0,0,0,0,0, 1.f);
  }

  // xf = LN_f(x)  (fp32 for probs, bf16 for lm_head)
  layernorm_kernel<<<BT_, 256, 0, stream>>>(x, lnf_g, lnf_b, xf_bf, xf);

  // logits = xf @ lm_w + lm_b  (written directly into d_out)
  const int blkW_DV = (int)(((long)D_ * V_ + 255) / 256);
  convt_kernel<<<blkW_DV, 256, 0, stream>>>(lm_w, wT, D_, V_);
  launch_gemm<EPI_BIAS_F32, 4>(stream, xf_bf, wT, logits, nullptr, lm_b,
                               nullptr, BT_, V_, D_, D_, D_, V_, 1, 1,
                               0,0,0,0,0,0, 1.f);

  // loss = mean cross-entropy over B*T rows
  hipMemsetAsync(loss, 0, sizeof(float), stream);
  loss_kernel<<<BT_, 256, 0, stream>>>(logits, targets, loss, V_,
                                       1.0f / (float)BT_);

  // probs = log_softmax(xf, axis=-1)
  logsoftmax_rows_kernel<<<BT_, 256, 0, stream>>>(xf, probs, D_);
}
